// S4Encoder_26998164423381
// MI455X (gfx1250) — matrix operand order repaired and checked by reading, not compiled, not run on hardware
//
#include <hip/hip_runtime.h>
#include <hip/hip_bf16.h>
#include <math.h>

#define B_ 64
#define H_ 256
#define L_ 1024
#define N_ 64
#define BH_ (B_ * H_)
#define BHL_ (B_ * H_ * L_)
#define HN_ (H_ * N_)

typedef __attribute__((ext_vector_type(16))) _Float16 v16h;
typedef __attribute__((ext_vector_type(8)))  _Float16 v8h;
typedef __attribute__((ext_vector_type(2)))  _Float16 h2;
typedef __attribute__((ext_vector_type(8)))  float    v8f;

union V16U { v16h v; v8h h[2]; };

// ---------------------------------------------------------------------------
// WMMA fragment loaders (v_wmma_f32_16x16x32_f16 operand layouts, ISA 7.12.2)
// A 16x32 f16: lanes 0-15 row M=lane, K{0..7,16..23}; lanes 16-31 K{8..15,24..31}
// B 32x16 f16: lanes 0-15 col N=lane, K{0..7,16..23}; lanes 16-31 K{8..15,24..31}
// ---------------------------------------------------------------------------
__device__ __forceinline__ v16h load_afrag(const _Float16* W, int stride, int mbase,
                                           int k0, int lane) {
  int row  = mbase + (lane & 15);
  int koff = (lane >> 4) ? 8 : 0;
  const _Float16* p = W + (size_t)row * stride + k0 + koff;
  V16U u;
  u.h[0] = *(const v8h*)p;
  u.h[1] = *(const v8h*)(p + 16);
  return u.v;
}

__device__ __forceinline__ v16h load_bfrag(const _Float16* T, int stride, int nrow,
                                           int k0, int lane) {
  int n    = nrow + (lane & 15);
  int koff = (lane >> 4) ? 8 : 0;
  const _Float16* p = T + (size_t)n * stride + k0 + koff;
  V16U u;
  u.h[0] = *(const v8h*)p;
  u.h[1] = *(const v8h*)(p + 16);
  return u.v;
}

__device__ __forceinline__ v8f wmma16(v16h a, v16h b, v8f c) {
  return __builtin_amdgcn_wmma_f32_16x16x32_f16(false, a, false, b, (short)0, c,
                                                false, false);
}

// ---------------------------------------------------------------------------
// Kernel 1a: timestep embedding -> silu, per batch.  V[b][i], i in [0,1024)
// ---------------------------------------------------------------------------
__global__ void k_emb_sin(const int* __restrict__ t, float* __restrict__ V) {
  int b = blockIdx.x;
  float tb = (float)t[b];
  const float cexp = -logf(10000.0f) / 511.0f;   // half-1 = 511
  for (int r = 0; r < 4; ++r) {
    int i = threadIdx.x + 256 * r;
    int k = (i < 512) ? i : (i - 512);
    float e = tb * __expf(cexp * (float)k);
    float v = (i < 512) ? sinf(e) : cosf(e);
    v = v / (1.0f + __expf(-v));                 // silu
    V[b * 1024 + i] = v;
  }
}

// ---------------------------------------------------------------------------
// Kernel 1b: emb2[b,o] = sum_k V[b,k] * ada_w[o,k] + ada_b[o]
// 8 blocks x 256 threads; thread owns one row o for all 64 batches so ada_w
// (8MB) streams from HBM exactly once total.
// ---------------------------------------------------------------------------
__global__ void k_emb_mm(const float* __restrict__ V, const float* __restrict__ ada_w,
                         const float* __restrict__ ada_b,
                         float* __restrict__ scaleA, float* __restrict__ shiftA) {
  __shared__ float Vs[64][65];
  int o = blockIdx.x * 256 + threadIdx.x;        // 0..2047
  float acc[64];
#pragma unroll
  for (int bb = 0; bb < 64; ++bb) acc[bb] = 0.f;
  for (int kc = 0; kc < 1024; kc += 64) {
    __syncthreads();
    for (int i = threadIdx.x; i < 64 * 64; i += 256) {
      int bb = i >> 6, kk = i & 63;
      Vs[bb][kk] = V[bb * 1024 + kc + kk];
    }
    __syncthreads();
    for (int kk = 0; kk < 64; ++kk) {
      float w = ada_w[(size_t)o * 1024 + kc + kk];
#pragma unroll
      for (int bb = 0; bb < 64; ++bb) acc[bb] = fmaf(w, Vs[bb][kk], acc[bb]);
    }
  }
  float bias = ada_b[o];
  for (int bb = 0; bb < 64; ++bb) {
    float val = acc[bb] + bias;
    if (o < 1024) scaleA[bb * 1024 + o] = val;
    else          shiftA[bb * 1024 + (o - 1024)] = val;
  }
}

// ---------------------------------------------------------------------------
// Kernel 2: AdaLayerNorm over L:  xa = ln(x)*(1+scale) + shift   (one row/blk)
// ---------------------------------------------------------------------------
__global__ void k_adaln(const float* __restrict__ x, const float* __restrict__ scaleA,
                        const float* __restrict__ shiftA, float* __restrict__ xa) {
  int row = blockIdx.x;                          // b*256 + h
  int b = row >> 8;
  const float* xr = x + (size_t)row * 1024;
  float v[4]; float s = 0.f, q = 0.f;
#pragma unroll
  for (int r = 0; r < 4; ++r) {
    v[r] = xr[threadIdx.x + 256 * r];
    s += v[r]; q += v[r] * v[r];
  }
#pragma unroll
  for (int m = 16; m >= 1; m >>= 1) { s += __shfl_xor(s, m, 32); q += __shfl_xor(q, m, 32); }
  __shared__ float ls[8], lq[8];
  int wave = threadIdx.x >> 5, lane = threadIdx.x & 31;
  if (lane == 0) { ls[wave] = s; lq[wave] = q; }
  __syncthreads();
  s = 0.f; q = 0.f;
#pragma unroll
  for (int w = 0; w < 8; ++w) { s += ls[w]; q += lq[w]; }
  float mean = s * (1.f / 1024.f);
  float var  = q * (1.f / 1024.f) - mean * mean;
  float rstd = rsqrtf(var + 1e-5f);
  float* xar = xa + (size_t)row * 1024;
  const float* sc = scaleA + (size_t)b * 1024;
  const float* sh = shiftA + (size_t)b * 1024;
#pragma unroll
  for (int r = 0; r < 4; ++r) {
    int l = threadIdx.x + 256 * r;
    xar[l] = (v[r] - mean) * rstd * (1.f + sc[l]) + sh[l];
  }
}

// ---------------------------------------------------------------------------
// Kernel 3: prenorm LayerNorm over H (channels), affine -> zh (f16).
// ---------------------------------------------------------------------------
__global__ void k_prenorm(const float* __restrict__ xa, const float* __restrict__ nw,
                          const float* __restrict__ nb, _Float16* __restrict__ zh) {
  __shared__ float tile[256][65];
  __shared__ float rs[4][64], rq[4][64];
  __shared__ float mA[64], rA[64];
  int b  = blockIdx.y;
  int l0 = blockIdx.x * 64;
  int lx = threadIdx.x & 63, hy = threadIdx.x >> 6;
  float s = 0.f, q = 0.f;
  for (int h = hy; h < 256; h += 4) {
    float v = xa[((size_t)(b * 256 + h)) * 1024 + l0 + lx];
    tile[h][lx] = v; s += v; q += v * v;
  }
  rs[hy][lx] = s; rq[hy][lx] = q;
  __syncthreads();
  if (hy == 0) {
    s = rs[0][lx] + rs[1][lx] + rs[2][lx] + rs[3][lx];
    q = rq[0][lx] + rq[1][lx] + rq[2][lx] + rq[3][lx];
    float m = s * (1.f / 256.f);
    float var = q * (1.f / 256.f) - m * m;
    mA[lx] = m; rA[lx] = rsqrtf(var + 1e-5f);
  }
  __syncthreads();
  float m = mA[lx], r = rA[lx];
  for (int h = hy; h < 256; h += 4) {
    float v = (tile[h][lx] - m) * r * nw[h] + nb[h];
    zh[((size_t)(b * 256 + h)) * 1024 + l0 + lx] = (_Float16)v;
  }
}

// ---------------------------------------------------------------------------
// Kernel 4: build per-h blocked-scan matrices (f16) + dA^Q (f32).
// Per-h f16 block (40960 halves): [T0 64x64][T1 64x64][Vf 128x64][Vb 128x64]
//                                 [Gf 64x128][Gb 64x128]
//  T0[j][i] = K0[j-i] (i<=j)              (intra fwd, lower-tri Toeplitz)
//  T1[j][i] = K1[i-j-1] (i>j)             (intra bwd, upper-tri Toeplitz)
//  Vf[2n][j]=Re dA^{63-j}, Vf[2n+1][j]=Im (U[c] = sum_j dA^{Q-1-j} z)
//  Vb[2n][j]=Re dA^{j},    Vb[2n+1][j]=Im (E[c] = sum_j dA^{j} z)
//  Gf[j][2n]=2Re(C0 dA^{j+1}),  Gf[j][2n+1]=-2Im(C0 dA^{j+1})
//  Gb[j][2n]=2Re(C1 dA^{63-j}), Gb[j][2n+1]=-2Im(C1 dA^{63-j})
// ---------------------------------------------------------------------------
__global__ void k_ssmprep(const float* __restrict__ log_dt, const float* __restrict__ A_re,
                          const float* __restrict__ A_im, const float* __restrict__ C_re,
                          const float* __restrict__ C_im,
                          _Float16* __restrict__ M16, float* __restrict__ dAQ) {
  __shared__ float dtAr[64], dtAi[64], c0r[64], c0i[64], c1r[64], c1i[64];
  __shared__ float K0s[64], K1s[64];
  int h = blockIdx.x;
  int tid = threadIdx.x;
  if (tid < 64) {
    int n = tid, idx = h * 64 + n;
    float dt = __expf(log_dt[h]);
    float Ar = -__expf(A_re[idx]);
    float Ai = A_im[idx];
    float ar = dt * Ar, ai = dt * Ai;
    dtAr[n] = ar; dtAi[n] = ai;
    float er = __expf(ar); float sn, cs; __sincosf(ai, &sn, &cs);
    float dr = er * cs, di = er * sn;
    float inv = 1.f / (Ar * Ar + Ai * Ai);
    float nr = dr - 1.f, ni = di;
    float qr = (nr * Ar + ni * Ai) * inv;        // (dA-1)/A
    float qi = (ni * Ar - nr * Ai) * inv;
    c0r[n] = C_re[idx] * qr - C_im[idx] * qi;
    c0i[n] = C_re[idx] * qi + C_im[idx] * qr;
    c1r[n] = C_re[HN_ + idx] * qr - C_im[HN_ + idx] * qi;
    c1i[n] = C_re[HN_ + idx] * qi + C_im[HN_ + idx] * qr;
    float eQ = __expf(64.f * ar); __sincosf(64.f * ai, &sn, &cs);
    dAQ[2 * idx] = eQ * cs; dAQ[2 * idx + 1] = eQ * sn;
  }
  __syncthreads();
  if (tid < 64) {                                // K0[d], K1[d]
    int d = tid;
    float k0 = 0.f, k1 = 0.f;
    for (int n = 0; n < 64; ++n) {
      float er = __expf((float)d * dtAr[n]);
      float sn, cs; __sincosf((float)d * dtAi[n], &sn, &cs);
      float wr = er * cs, wi = er * sn;
      k0 += c0r[n] * wr - c0i[n] * wi;
      k1 += c1r[n] * wr - c1i[n] * wi;
    }
    K0s[d] = 2.f * k0; K1s[d] = 2.f * k1;
  }
  __syncthreads();
  _Float16* Mh = M16 + (size_t)h * 40960;
  _Float16* T0 = Mh;
  _Float16* T1 = Mh + 4096;
  _Float16* Vf = Mh + 8192;
  _Float16* Vb = Mh + 16384;
  _Float16* Gf = Mh + 24576;
  _Float16* Gb = Mh + 32768;
  for (int i = tid; i < 4096; i += 256) {
    int j = i >> 6, ii = i & 63;
    T0[i] = (_Float16)((ii <= j) ? K0s[j - ii] : 0.f);
    T1[i] = (_Float16)((ii > j) ? K1s[ii - j - 1] : 0.f);
  }
  for (int i = tid; i < 4096; i += 256) {
    int n = i >> 6, j = i & 63;
    float sn, cs;
    float df = (float)(63 - j);
    float er = __expf(df * dtAr[n]); __sincosf(df * dtAi[n], &sn, &cs);
    Vf[(2 * n) * 64 + j]     = (_Float16)(er * cs);
    Vf[(2 * n + 1) * 64 + j] = (_Float16)(er * sn);
    float db = (float)j;
    er = __expf(db * dtAr[n]); __sincosf(db * dtAi[n], &sn, &cs);
    Vb[(2 * n) * 64 + j]     = (_Float16)(er * cs);
    Vb[(2 * n + 1) * 64 + j] = (_Float16)(er * sn);
  }
  for (int i = tid; i < 4096; i += 256) {
    int j = i >> 6, n = i & 63;
    float sn, cs;
    float df = (float)(j + 1);
    float er = __expf(df * dtAr[n]); __sincosf(df * dtAi[n], &sn, &cs);
    float wr = er * cs, wi = er * sn;
    Gf[j * 128 + 2 * n]     = (_Float16)( 2.f * (c0r[n] * wr - c0i[n] * wi));
    Gf[j * 128 + 2 * n + 1] = (_Float16)(-2.f * (c0r[n] * wi + c0i[n] * wr));
    float db = (float)(63 - j);
    er = __expf(db * dtAr[n]); __sincosf(db * dtAi[n], &sn, &cs);
    wr = er * cs; wi = er * sn;
    Gb[j * 128 + 2 * n]     = (_Float16)( 2.f * (c1r[n] * wr - c1i[n] * wi));
    Gb[j * 128 + 2 * n + 1] = (_Float16)(-2.f * (c1r[n] * wi + c1i[n] * wr));
  }
}

// ---------------------------------------------------------------------------
// Kernel 5: blocked bidirectional scan, one workgroup per (b,h) sequence.
//   stage A (WMMA): U = Vf x Z, E = Vb x Z             (chunk->state)
//   stage B (VALU): 16-step chunk scans  S[c]=dA^Q S[c-1]+U[c] (and reverse)
//   stage C (WMMA): Y = T0 Z + Gf S_prev + T1 Z + Gb R_next
//   epilogue: y += D*z, exact gelu, write f16 yg.
// Z tile staged via ASYNC global->LDS copy (ASYNCcnt).
// ---------------------------------------------------------------------------
__global__ void __launch_bounds__(256) k_scan(const _Float16* __restrict__ zh,
                                              const _Float16* __restrict__ M16,
                                              const float* __restrict__ dAQ,
                                              const float* __restrict__ Dp,
                                              _Float16* __restrict__ yg) {
  __shared__ _Float16 Zls[16][64];               // [chunk c][j]
  __shared__ float    Uls[128][17];
  __shared__ float    Els[128][17];
  __shared__ _Float16 Sls[16][136];              // S[c-1], interleaved re/im
  __shared__ _Float16 Rls[16][136];              // R[c+1]
  __shared__ float    Yp[4][8][32];              // partner partial acc
  int seq = blockIdx.x;                          // b*256 + h
  int h = seq & 255;
  int tid = threadIdx.x;
  int wave = tid >> 5, lane = tid & 31;
  const _Float16* Mh = M16 + (size_t)h * 40960;
  const _Float16* T0 = Mh;
  const _Float16* T1 = Mh + 4096;
  const _Float16* Vf = Mh + 8192;
  const _Float16* Vb = Mh + 16384;
  const _Float16* Gf = Mh + 24576;
  const _Float16* Gb = Mh + 32768;
  const _Float16* zrow = zh + (size_t)seq * 1024;

  // --- async copy of the 2KB z tile into LDS (tracked by ASYNCcnt) ---
  if (tid < 128) {
    const _Float16* gsrc = zrow + tid * 8;       // 16B per lane
    unsigned lds_off = (unsigned)(size_t)(&Zls[0][0]) + tid * 16;
    asm volatile("global_load_async_to_lds_b128 %0, %1, off"
                 :: "v"(lds_off), "v"(gsrc) : "memory");
  }
  asm volatile("s_wait_asynccnt 0" ::: "memory");
  __syncthreads();

  const v8f zero = {0.f, 0.f, 0.f, 0.f, 0.f, 0.f, 0.f, 0.f};

  // --- stage A: each wave computes one 16-row tile of U and of E ---
  {
    v8f au = zero, ae = zero;
    for (int kt = 0; kt < 2; ++kt) {
      int k0 = kt * 32;
      v16h b  = load_bfrag(&Zls[0][0], 64, 0, k0, lane);
      v16h af = load_afrag(Vf, 64, wave * 16, k0, lane);
      v16h ab = load_afrag(Vb, 64, wave * 16, k0, lane);
      au = wmma16(af, b, au);
      ae = wmma16(ab, b, ae);
    }
    int row0 = wave * 16 + ((lane >> 4) ? 8 : 0);
    int c = lane & 15;
#pragma unroll
    for (int r = 0; r < 8; ++r) { Uls[row0 + r][c] = au[r]; Els[row0 + r][c] = ae[r]; }
  }
  __syncthreads();

  // --- stage B: 16-step chunk scans (fwd: threads 0-63, bwd: 64-127) ---
  if (tid < 64) {
    int n = tid;
    float aQr = dAQ[2 * (h * 64 + n)], aQi = dAQ[2 * (h * 64 + n) + 1];
    float sr = 0.f, si = 0.f;
    for (int c = 0; c < 16; ++c) {
      Sls[c][2 * n]     = (_Float16)sr;          // S[c-1]
      Sls[c][2 * n + 1] = (_Float16)si;
      float ur = Uls[2 * n][c], ui = Uls[2 * n + 1][c];
      float nsr = aQr * sr - aQi * si + ur;
      si = aQr * si + aQi * sr + ui; sr = nsr;
    }
  } else if (tid < 128) {
    int n = tid - 64;
    float aQr = dAQ[2 * (h * 64 + n)], aQi = dAQ[2 * (h * 64 + n) + 1];
    float rr = 0.f, ri = 0.f;
    for (int c = 15; c >= 0; --c) {
      Rls[c][2 * n]     = (_Float16)rr;          // R[c+1]
      Rls[c][2 * n + 1] = (_Float16)ri;
      float er = Els[2 * n][c], ei = Els[2 * n + 1][c];
      float nrr = aQr * rr - aQi * ri + er;
      ri = aQr * ri + aQi * rr + ei; rr = nrr;
    }
  }
  __syncthreads();

  // --- stage C: waves 0-3: T0*Z + Gf*S ; waves 4-7: T1*Z + Gb*R ---
  {
    int m = wave & 3;                            // j-tile: rows 16m..16m+15
    v8f acc = zero;
    if (wave < 4) {
      for (int kt = 0; kt < 2; ++kt) {
        v16h a = load_afrag(T0, 64, m * 16, kt * 32, lane);
        v16h b = load_bfrag(&Zls[0][0], 64, 0, kt * 32, lane);
        acc = wmma16(a, b, acc);
      }
      for (int kt = 0; kt < 4; ++kt) {
        v16h a = load_afrag(Gf, 128, m * 16, kt * 32, lane);
        v16h b = load_bfrag(&Sls[0][0], 136, 0, kt * 32, lane);
        acc = wmma16(a, b, acc);
      }
    } else {
      for (int kt = 0; kt < 2; ++kt) {
        v16h a = load_afrag(T1, 64, m * 16, kt * 32, lane);
        v16h b = load_bfrag(&Zls[0][0], 64, 0, kt * 32, lane);
        acc = wmma16(a, b, acc);
      }
      for (int kt = 0; kt < 4; ++kt) {
        v16h a = load_afrag(Gb, 128, m * 16, kt * 32, lane);
        v16h b = load_bfrag(&Rls[0][0], 136, 0, kt * 32, lane);
        acc = wmma16(a, b, acc);
      }
#pragma unroll
      for (int r = 0; r < 8; ++r) Yp[m][r][lane] = acc[r];
    }
    __syncthreads();
    if (wave < 4) {
      float Dh = Dp[h];
      _Float16* ygr = yg + (size_t)seq * 1024;
      int j0 = m * 16 + ((lane >> 4) ? 8 : 0);
      int c  = lane & 15;
      v8h out;
#pragma unroll
      for (int r = 0; r < 8; ++r) {
        int j = j0 + r;
        float zv = (float)Zls[c][j];
        float yt = acc[r] + Yp[m][r][lane] + Dh * zv;
        out[r] = (_Float16)(0.5f * yt * (1.f + erff(yt * 0.70710678118654752f)));
      }
      *(v8h*)(ygr + c * 64 + j0) = out;          // 8 contiguous f16
    }
  }
}

// ---------------------------------------------------------------------------
// Kernel 6: cast the three HxH weight matrices to f16 for WMMA
// ---------------------------------------------------------------------------
__global__ void k_castw(const float* __restrict__ w0, const float* __restrict__ w1,
                        const float* __restrict__ w2, _Float16* __restrict__ w16) {
  int i = blockIdx.x * 256 + threadIdx.x;
  if (i < 65536)            w16[i] = (_Float16)w0[i];
  else if (i < 2 * 65536)   w16[i] = (_Float16)w1[i - 65536];
  else                      w16[i] = (_Float16)w2[i - 2 * 65536];
}

// ---------------------------------------------------------------------------
// Kernel 7: fused channel-mixing (unchanged from round 1).
// ---------------------------------------------------------------------------
__global__ void __launch_bounds__(256) k_mix(const _Float16* __restrict__ yg,
                                             const float* __restrict__ xa,
                                             const _Float16* __restrict__ w16,
                                             const float* __restrict__ out_b,
                                             const float* __restrict__ lin1_b,
                                             const float* __restrict__ lin2_b,
                                             float* __restrict__ out1,
                                             float* __restrict__ out2) {
  __shared__ _Float16 T[64][264];                // act tile, transposed [l][k]
  int b  = blockIdx.y;
  int l0 = blockIdx.x * 64;
  int tid = threadIdx.x;
  int wave = tid >> 5, lane = tid & 31;
  const _Float16* W0 = w16;
  const _Float16* W1 = w16 + 65536;
  const _Float16* W2 = w16 + 2 * 65536;
  const float* xab = xa + ((size_t)b * 256) * 1024;
  __builtin_prefetch(W1 + (size_t)(wave * 32) * 256, 0, 3);   // global_prefetch_b8
  __builtin_prefetch(W2 + (size_t)(wave * 32) * 256, 0, 3);

  {
    const _Float16* ygb = yg + ((size_t)b * 256) * 1024 + l0;
    for (int i = tid; i < 8192; i += 256) {
      int hh = i >> 5;
      int lp = i & 31;
      h2 v = *(const h2*)(ygb + (size_t)hh * 1024 + lp * 2);
      T[lp * 2][hh]     = v.x;
      T[lp * 2 + 1][hh] = v.y;
    }
  }
  __syncthreads();

  const v8f zero = {0.f, 0.f, 0.f, 0.f, 0.f, 0.f, 0.f, 0.f};
  int mbase = wave * 32;
  v8f acc[2][4];

  // ---- GEMM1: out_w @ yg ----
#pragma unroll
  for (int mi = 0; mi < 2; ++mi)
#pragma unroll
    for (int ni = 0; ni < 4; ++ni) acc[mi][ni] = zero;
  for (int kt = 0; kt < 8; ++kt) {
    int k0 = kt * 32;
    v16h a0 = load_afrag(W0, 256, mbase, k0, lane);
    v16h a1 = load_afrag(W0, 256, mbase + 16, k0, lane);
#pragma unroll
    for (int ni = 0; ni < 4; ++ni) {
      v16h bf = load_bfrag(&T[0][0], 264, ni * 16, k0, lane);
      acc[0][ni] = wmma16(a0, bf, acc[0][ni]);
      acc[1][ni] = wmma16(a1, bf, acc[1][ni]);
    }
  }
  __syncthreads();

  // epilogue 1: bias + residual + tanh*sigmoid gate -> T[l][m] (f16)
#pragma unroll
  for (int mi = 0; mi < 2; ++mi) {
    int m0 = mbase + mi * 16 + ((lane >> 4) ? 8 : 0);
#pragma unroll
    for (int ni = 0; ni < 4; ++ni) {
      int l = l0 + ni * 16 + (lane & 15);
      v8h gh;
#pragma unroll
      for (int r = 0; r < 8; ++r) {
        int m = m0 + r;
        float y2 = acc[mi][ni][r] + out_b[m];
        float x2 = y2 + xab[(size_t)m * 1024 + l];
        float g  = tanhf(x2) * (1.f / (1.f + __expf(-x2)));
        gh[r] = (_Float16)g;
      }
      *(v8h*)&T[ni * 16 + (lane & 15)][m0] = gh;
    }
  }
  __syncthreads();

  // ---- GEMM2: lin1_w @ g -> out1 (+bias +xa residual) ----
#pragma unroll
  for (int mi = 0; mi < 2; ++mi)
#pragma unroll
    for (int ni = 0; ni < 4; ++ni) acc[mi][ni] = zero;
  for (int kt = 0; kt < 8; ++kt) {
    int k0 = kt * 32;
    v16h a0 = load_afrag(W1, 256, mbase, k0, lane);
    v16h a1 = load_afrag(W1, 256, mbase + 16, k0, lane);
#pragma unroll
    for (int ni = 0; ni < 4; ++ni) {
      v16h bf = load_bfrag(&T[0][0], 264, ni * 16, k0, lane);
      acc[0][ni] = wmma16(a0, bf, acc[0][ni]);
      acc[1][ni] = wmma16(a1, bf, acc[1][ni]);
    }
  }
#pragma unroll
  for (int mi = 0; mi < 2; ++mi) {
    int m0 = mbase + mi * 16 + ((lane >> 4) ? 8 : 0);
#pragma unroll
    for (int ni = 0; ni < 4; ++ni) {
      int l = l0 + ni * 16 + (lane & 15);
#pragma unroll
      for (int r = 0; r < 8; ++r) {
        int m = m0 + r;
        out1[((size_t)(b * 256 + m)) * 1024 + l] =
            acc[mi][ni][r] + lin1_b[m] + xab[(size_t)m * 1024 + l];
      }
    }
  }

  // ---- GEMM3: lin2_w @ g -> out2 (+bias) ----
#pragma unroll
  for (int mi = 0; mi < 2; ++mi)
#pragma unroll
    for (int ni = 0; ni < 4; ++ni) acc[mi][ni] = zero;
  for (int kt = 0; kt < 8; ++kt) {
    int k0 = kt * 32;
    v16h a0 = load_afrag(W2, 256, mbase, k0, lane);
    v16h a1 = load_afrag(W2, 256, mbase + 16, k0, lane);
#pragma unroll
    for (int ni = 0; ni < 4; ++ni) {
      v16h bf = load_bfrag(&T[0][0], 264, ni * 16, k0, lane);
      acc[0][ni] = wmma16(a0, bf, acc[0][ni]);
      acc[1][ni] = wmma16(a1, bf, acc[1][ni]);
    }
  }
#pragma unroll
  for (int mi = 0; mi < 2; ++mi) {
    int m0 = mbase + mi * 16 + ((lane >> 4) ? 8 : 0);
#pragma unroll
    for (int ni = 0; ni < 4; ++ni) {
      int l = l0 + ni * 16 + (lane & 15);
#pragma unroll
      for (int r = 0; r < 8; ++r) {
        int m = m0 + r;
        out2[((size_t)(b * 256 + m)) * 1024 + l] = acc[mi][ni][r] + lin2_b[m];
      }
    }
  }
}

// ---------------------------------------------------------------------------
extern "C" void kernel_launch(void* const* d_in, const int* in_sizes, int n_in,
                              void* d_out, int out_size, void* d_ws, size_t ws_size,
                              hipStream_t stream) {
  (void)in_sizes; (void)n_in; (void)out_size; (void)ws_size;
  const float* x      = (const float*)d_in[0];
  const int*   t      = (const int*)d_in[1];
  const float* ada_w  = (const float*)d_in[2];
  const float* ada_b  = (const float*)d_in[3];
  const float* norm_w = (const float*)d_in[4];
  const float* norm_b = (const float*)d_in[5];
  const float* log_dt = (const float*)d_in[6];
  const float* A_re   = (const float*)d_in[7];
  const float* A_im   = (const float*)d_in[8];
  const float* C_re   = (const float*)d_in[9];
  const float* C_im   = (const float*)d_in[10];
  const float* Dp     = (const float*)d_in[11];
  const float* out_w  = (const float*)d_in[12];
  const float* out_b  = (const float*)d_in[13];
  const float* lin1_w = (const float*)d_in[14];
  const float* lin1_b = (const float*)d_in[15];
  const float* lin2_w = (const float*)d_in[16];
  const float* lin2_b = (const float*)d_in[17];

  char* ws = (char*)d_ws;
  float*    V      = (float*)(ws);                                        // 256 KB
  float*    scaleA = (float*)(ws + (256u << 10));                         // 256 KB
  float*    shiftA = (float*)(ws + (512u << 10));                         // 256 KB
  float*    xa     = (float*)(ws + (768u << 10));                         // 64 MB
  _Float16* zh     = (_Float16*)(ws + (768u << 10) + (64u << 20));        // 32 MB
  _Float16* M16    = (_Float16*)(ws + (768u << 10) + (96u << 20));        // 20 MB
  float*    dAQ    = (float*)(ws + (768u << 10) + (116u << 20));          // 128 KB
  _Float16* yg     = (_Float16*)(ws + (768u << 10) + (117u << 20));       // 32 MB
  _Float16* w16    = (_Float16*)(ws + (768u << 10) + (149u << 20));       // 384 KB

  float* out1 = (float*)d_out;
  float* out2 = out1 + (size_t)BHL_;

  k_emb_sin<<<B_, 256, 0, stream>>>(t, V);
  k_emb_mm <<<8, 256, 0, stream>>>(V, ada_w, ada_b, scaleA, shiftA);
  k_adaln  <<<BH_, 256, 0, stream>>>(x, scaleA, shiftA, xa);
  k_prenorm<<<dim3(16, 64), 256, 0, stream>>>(xa, norm_w, norm_b, zh);
  k_ssmprep<<<H_, 256, 0, stream>>>(log_dt, A_re, A_im, C_re, C_im, M16, dAQ);
  k_scan   <<<BH_, 256, 0, stream>>>(zh, M16, dAQ, Dp, yg);
  k_castw  <<<768, 256, 0, stream>>>(out_w, lin1_w, lin2_w, w16);
  k_mix    <<<dim3(16, 64), 256, 0, stream>>>(yg, xa, w16, out_b, lin1_b, lin2_b, out1, out2);
}
